// SCoNe_1760936591462
// MI455X (gfx1250) — matrix operand order repaired and checked by reading, not compiled, not run on hardware
//
#include <hip/hip_runtime.h>
#include <hip/hip_bf16.h>

#define N_NODES 50000
#define N_EDGES 150000
#define N_TRIS  100000
#define FDIM    256
#define NLAYERS 4

// GEMM blocking: 80 edges (5 M-tiles of 16) per block, 20 waves (640 threads).
// 150000 / 80 = 1875 blocks exactly (no tail).
#define MT_PER_BLK 5
#define M_BLK      (MT_PER_BLK * 16)
#define GEMM_THREADS (MT_PER_BLK * 4 * 32)   // (mt, nt-group) wave grid

typedef __attribute__((ext_vector_type(16))) __bf16         v16bf;
typedef __attribute__((ext_vector_type(16))) unsigned short v16us;
typedef __attribute__((ext_vector_type(8)))  float          v8f;
typedef __attribute__((ext_vector_type(4)))  float          f4;

union FragU { v16us u; v16bf b; };

__device__ __forceinline__ unsigned short f2bf(float f) {
  unsigned u = __float_as_uint(f);
  u += 0x7FFFu + ((u >> 16) & 1u);          // round-to-nearest-even
  return (unsigned short)(u >> 16);
}

// ---------------------------------------------------------------------------
// Convert W0/W1/W2 stacks (L x 256 x 256 fp32) into bf16 WMMA B-fragments,
// fragment-major: frag = ((w*L + layer)*8 + kk)*16 + nt, 32 lanes x 16 halfs.
// B layout (32x16, 16-bit): lane holds N = lane&15; lane-half g selects K offset
// 8g; halfs 0-7 -> K 0-7(+8g), halfs 8-15 -> K 16-23(+8g).
// ---------------------------------------------------------------------------
__global__ __launch_bounds__(256) void k_wconv(
    const float* __restrict__ W0, const float* __restrict__ W1,
    const float* __restrict__ W2, unsigned short* __restrict__ wperm) {
  const int tid  = blockIdx.x * 256 + threadIdx.x;       // one half-word each
  const int frag = tid >> 9;                              // 512 halfs / frag
  const int r    = tid & 511;
  const int l    = r >> 4;
  const int h    = r & 15;
  const int nt    = frag & 15;
  const int kk    = (frag >> 4) & 7;
  const int layer = (frag >> 7) & 3;
  const int w     = frag >> 9;                            // 0..2
  const int k = kk * 32 + ((l >> 4) << 3) + (h & 7) + ((h >> 3) << 4);
  const int n = nt * 16 + (l & 15);
  const float* Ws = (w == 0) ? W0 : ((w == 1) ? W1 : W2);
  wperm[tid] = f2bf(Ws[(size_t)layer * FDIM * FDIM + (size_t)k * FDIM + n]);
}

// ---------------------------------------------------------------------------
// Fused triple GEMM: y0 = x@W0, y1 = x@W1, y2 = x@W2 for one 80-edge tile.
// 20 waves/block: wave = (mt, nt-group); each wave owns one 16-row M tile and
// a 64-wide N slice for all three weights (12 accumulator tiles, 96 WMMAs).
// Weight fragments are read once per block -> 5x less L2 weight traffic than
// a 16-edge block.
// ---------------------------------------------------------------------------
__global__ __launch_bounds__(GEMM_THREADS) void k_gemm3(
    const float* __restrict__ xin, const unsigned short* __restrict__ wperm,
    int layer, float* __restrict__ y0, float* __restrict__ y1,
    float* __restrict__ y2) {
  // A fragments, frag-major: [mt][kk][lane] x 32B  = 40 KB
  __shared__ __align__(32) unsigned short aLds[MT_PER_BLK * 8 * 32 * 16];
  const int t    = threadIdx.x;
  const int lane = t & 31;
  const int wave = t >> 5;
  const int mt   = wave >> 2;                // 0..4
  const int ntg  = wave & 3;                 // 0..3 -> nt = ntg*4 + j
  const int m0   = blockIdx.x * M_BLK;

  // Stage x tile as bf16 A-fragments (A 16x32 layout: M = lane&15; lane-half g
  // selects K offset 8g; halfs 0-7 -> K 0-7(+8g), halfs 8-15 -> K 16-23(+8g)).
  for (int a = t; a < MT_PER_BLK * 8 * 32; a += GEMM_THREADS) {
    const int amt = a >> 8;                  // 8*32 frag-lanes per M tile
    const int kk  = (a >> 5) & 7;
    const int l   = a & 31;
    const int g = (l >> 4) * 8, m = (l & 15) + amt * 16;
    const float* src = xin + (size_t)(m0 + m) * FDIM + kk * 32 + g;
    __builtin_prefetch(src + (size_t)M_BLK * FDIM, 0, 0);  // next block, speculative
    f4 f0 = *(const f4*)(src + 0);
    f4 f1 = *(const f4*)(src + 4);
    f4 f2 = *(const f4*)(src + 16);
    f4 f3 = *(const f4*)(src + 20);
    FragU fr;
    fr.u[0]  = f2bf(f0[0]); fr.u[1]  = f2bf(f0[1]); fr.u[2]  = f2bf(f0[2]); fr.u[3]  = f2bf(f0[3]);
    fr.u[4]  = f2bf(f1[0]); fr.u[5]  = f2bf(f1[1]); fr.u[6]  = f2bf(f1[2]); fr.u[7]  = f2bf(f1[3]);
    fr.u[8]  = f2bf(f2[0]); fr.u[9]  = f2bf(f2[1]); fr.u[10] = f2bf(f2[2]); fr.u[11] = f2bf(f2[3]);
    fr.u[12] = f2bf(f3[0]); fr.u[13] = f2bf(f3[1]); fr.u[14] = f2bf(f3[2]); fr.u[15] = f2bf(f3[3]);
    ((v16us*)aLds)[a] = fr.u;
  }
  __syncthreads();

  v8f acc[3][4] = {};
  const v16us* wfr = (const v16us*)wperm;
#pragma unroll
  for (int kk = 0; kk < 8; ++kk) {
    FragU af;
    af.u = ((const v16us*)aLds)[(mt * 8 + kk) * 32 + lane];  // ds_load_b128 x2
#pragma unroll
    for (int w = 0; w < 3; ++w) {
      // Batch the 4 B-fragment loads so the scheduler can clause them and
      // overlap L2 latency with the WMMA chain (partial loadcnt waits).
      FragU bb[4];
#pragma unroll
      for (int j = 0; j < 4; ++j) {
        const int nt = ntg * 4 + j;
        bb[j].u = wfr[((size_t)((w * NLAYERS + layer) * 8 + kk) * 16 + nt) * 32 + lane];
      }
#pragma unroll
      for (int j = 0; j < 4; ++j)
        acc[w][j] = __builtin_amdgcn_wmma_f32_16x16x32_bf16(
            false, af.b, false, bb[j].b, (short)0, acc[w][j], false, false);
    }
  }

  // D layout: VGPR r -> row m = r + 8*(lane>>4), col n = lane&15.
  float* const ys[3] = {y0, y1, y2};
  const int n_lo = lane & 15;
  const int mb   = m0 + mt * 16 + ((lane >> 4) << 3);
#pragma unroll
  for (int w = 0; w < 3; ++w)
#pragma unroll
    for (int j = 0; j < 4; ++j) {
      const int n = (ntg * 4 + j) * 16 + n_lo;
#pragma unroll
      for (int r = 0; r < 8; ++r)
        ys[w][(size_t)(mb + r) * FDIM + n] = acc[w][j][r];
    }
}

// ---------------------------------------------------------------------------
// Incidence operators
// ---------------------------------------------------------------------------
__global__ __launch_bounds__(256) void k_zero(float* __restrict__ p, long n) {
  long i = (long)blockIdx.x * 256 + threadIdx.x;
  if (i < n) p[i] = 0.0f;
}

// z = B1 @ y0 : z[u] -= y0[e], z[v] += y0[e]
__global__ __launch_bounds__(256) void k_b1_scatter(
    const float* __restrict__ y, const int* __restrict__ en,
    float* __restrict__ z) {
  const int e = blockIdx.x, f = threadIdx.x;
  const float v  = y[(size_t)e * FDIM + f];
  const int  nu = en[2 * e], nv = en[2 * e + 1];
  atomicAdd(z + (size_t)nu * FDIM + f, -v);
  atomicAdd(z + (size_t)nv * FDIM + f,  v);
}

// wt = B2^T @ y2 : wt[t] = sum_k s[t,k] * y2[te[t,k]]
__global__ __launch_bounds__(256) void k_b2t_gather(
    const float* __restrict__ y, const int* __restrict__ te,
    const int* __restrict__ ts, float* __restrict__ wt) {
  const int tr = blockIdx.x, f = threadIdx.x;
  const int e0 = te[3 * tr], e1 = te[3 * tr + 1], e2 = te[3 * tr + 2];
  const float s0 = (float)ts[3 * tr], s1 = (float)ts[3 * tr + 1],
              s2 = (float)ts[3 * tr + 2];
  wt[(size_t)tr * FDIM + f] = s0 * y[(size_t)e0 * FDIM + f] +
                              s1 * y[(size_t)e1 * FDIM + f] +
                              s2 * y[(size_t)e2 * FDIM + f];
}

// d2 = B2 @ wt : d2[te[t,k]] += s[t,k] * wt[t]
__global__ __launch_bounds__(256) void k_b2_scatter(
    const float* __restrict__ wt, const int* __restrict__ te,
    const int* __restrict__ ts, float* __restrict__ d2) {
  const int tr = blockIdx.x, f = threadIdx.x;
  const float v = wt[(size_t)tr * FDIM + f];
#pragma unroll
  for (int k = 0; k < 3; ++k) {
    const int   e = te[3 * tr + k];
    const float s = (float)ts[3 * tr + k];
    atomicAdd(d2 + (size_t)e * FDIM + f, s * v);
  }
}

// x_next = relu(y1 + d2 + (z[v] - z[u]))
__global__ __launch_bounds__(256) void k_combine(
    const float* __restrict__ y1, const float* __restrict__ d2,
    const float* __restrict__ z, const int* __restrict__ en,
    float* __restrict__ xo) {
  const int e = blockIdx.x, f = threadIdx.x;
  const int nu = en[2 * e], nv = en[2 * e + 1];
  const float v = y1[(size_t)e * FDIM + f] + d2[(size_t)e * FDIM + f] +
                  z[(size_t)nv * FDIM + f] - z[(size_t)nu * FDIM + f];
  xo[(size_t)e * FDIM + f] = fmaxf(v, 0.0f);
}

// out = B1 @ (x @ W0_L): one wave per edge, lane-strided dot + shfl reduce.
__global__ __launch_bounds__(256) void k_final(
    const float* __restrict__ x, const float* __restrict__ w,
    const int* __restrict__ en, float* __restrict__ out) {
  const int lane = threadIdx.x & 31;
  const int e    = blockIdx.x * 8 + (threadIdx.x >> 5);
  const float* row = x + (size_t)e * FDIM;
  float s = 0.0f;
#pragma unroll
  for (int j = 0; j < 8; ++j) s += row[lane + 32 * j] * w[lane + 32 * j];
#pragma unroll
  for (int off = 16; off > 0; off >>= 1) s += __shfl_xor(s, off, 32);
  if (lane == 0) {
    atomicAdd(out + en[2 * e], -s);
    atomicAdd(out + en[2 * e + 1], s);
  }
}

// ---------------------------------------------------------------------------
extern "C" void kernel_launch(void* const* d_in, const int* in_sizes, int n_in,
                              void* d_out, int out_size, void* d_ws,
                              size_t ws_size, hipStream_t stream) {
  const float* x0  = (const float*)d_in[0];
  const float* W0  = (const float*)d_in[1];
  const float* W1  = (const float*)d_in[2];
  const float* W2  = (const float*)d_in[3];
  const float* W0L = (const float*)d_in[4];
  const int*   en  = (const int*)d_in[5];
  const int*   te  = (const int*)d_in[6];
  const int*   ts  = (const int*)d_in[7];
  float* out = (float*)d_out;

  const size_t EF = (size_t)N_EDGES * FDIM;
  float* xa = (float*)d_ws;
  float* xb = xa + EF;
  float* y0 = xb + EF;                       // reused as d2 accumulator
  float* y1 = y0 + EF;
  float* y2 = y1 + EF;
  float* z  = y2 + EF;                       // N_NODES * F
  float* wt = z + (size_t)N_NODES * FDIM;    // N_TRIS * F
  unsigned short* wperm = (unsigned short*)(wt + (size_t)N_TRIS * FDIM);

  // Deterministic per-call weight conversion (no caching allowed).
  k_wconv<<<(3 * NLAYERS * FDIM * FDIM) / 256, 256, 0, stream>>>(W0, W1, W2, wperm);

  const float* cur = x0;
  for (int i = 0; i < NLAYERS; ++i) {
    float* nxt = (i & 1) ? xb : xa;
    k_gemm3<<<N_EDGES / M_BLK, GEMM_THREADS, 0, stream>>>(cur, wperm, i,
                                                          y0, y1, y2);
    k_zero<<<(unsigned)(((size_t)N_NODES * FDIM + 255) / 256), 256, 0, stream>>>(
        z, (long)N_NODES * FDIM);
    k_b1_scatter<<<N_EDGES, 256, 0, stream>>>(y0, en, z);
    k_b2t_gather<<<N_TRIS, 256, 0, stream>>>(y2, te, ts, wt);
    k_zero<<<(unsigned)((EF + 255) / 256), 256, 0, stream>>>(y0, (long)EF);
    k_b2_scatter<<<N_TRIS, 256, 0, stream>>>(wt, te, ts, y0);
    k_combine<<<N_EDGES, 256, 0, stream>>>(y1, y0, z, en, nxt);
    cur = nxt;
  }

  k_zero<<<(N_NODES + 255) / 256, 256, 0, stream>>>(out, N_NODES);
  k_final<<<N_EDGES / 8, 256, 0, stream>>>(cur, W0L, en, out);
}
